// Model7_9620726743223
// MI455X (gfx1250) — matrix operand order repaired and checked by reading, not compiled, not run on hardware
//
#include <hip/hip_runtime.h>
#include <hip/hip_bf16.h>
#include <math.h>

typedef __attribute__((ext_vector_type(16))) _Float16 v16h;
typedef __attribute__((ext_vector_type(8)))  float    v8f;

#define NEGSLOPE 0.2f

__device__ __forceinline__ float leaky(float v) { return v > 0.f ? v : NEGSLOPE * v; }
// order-preserving float->uint encoding for atomicMax on floats
__device__ __forceinline__ unsigned fenc(float f) {
    unsigned u = __float_as_uint(f);
    return (u & 0x80000000u) ? ~u : (u | 0x80000000u);
}
__device__ __forceinline__ float fdec(unsigned k) {
    unsigned u = (k & 0x80000000u) ? (k ^ 0x80000000u) : ~k;
    return __uint_as_float(u);
}

// ---------------- generic fill ----------------
__global__ void k_fill(unsigned* __restrict__ p, unsigned v, int n) {
    int i = blockIdx.x * blockDim.x + threadIdx.x;
    if (i < n) p[i] = v;
}

// ---------------- layer-1 concat buffer: [N,32] = [x1(15), 0...] ----------------
__global__ void k_build_cat1(const float* __restrict__ x1, float* __restrict__ xcat, int N) {
    int n = blockIdx.x * blockDim.x + threadIdx.x;
    if (n >= N) return;
    float* c = xcat + (size_t)n * 32;
    const float* r = x1 + (size_t)n * 15;
#pragma unroll
    for (int f = 0; f < 15; ++f) c[f] = r[f];
#pragma unroll
    for (int f = 15; f < 32; ++f) c[f] = 0.f;
}

// load A fragment (16x32 f16) from xcat row (clamped; row-independence makes clamp exact)
__device__ __forceinline__ v16h load_afrag(const float* __restrict__ xcat, int row, int hi) {
    const float4* x4 = (const float4*)(xcat + (size_t)row * 32);
    int q = hi ? 4 : 0;                      // K base 0 -> f4 idx 0..3 / K base 16 -> 4..7
    float4 a0 = x4[q], a1 = x4[q + 1];       // K = kb .. kb+7
    float4 a2 = x4[q + 2], a3 = x4[q + 3];   // K = kb+8 .. kb+15
    v16h a;
    a[0] = (_Float16)a0.x;  a[1] = (_Float16)a0.y;  a[2]  = (_Float16)a0.z;  a[3]  = (_Float16)a0.w;
    a[4] = (_Float16)a1.x;  a[5] = (_Float16)a1.y;  a[6]  = (_Float16)a1.z;  a[7]  = (_Float16)a1.w;
    a[8] = (_Float16)a2.x;  a[9] = (_Float16)a2.y;  a[10] = (_Float16)a2.z;  a[11] = (_Float16)a2.w;
    a[12] = (_Float16)a3.x; a[13] = (_Float16)a3.y; a[14] = (_Float16)a3.z;  a[15] = (_Float16)a3.w;
    return a;
}

// ---------------- WMMA GEMM: xl = xcat @ Wl^T, xr = xcat @ Wr^T  (out 10 cols, padded to 16) ----
__global__ void k_gemm_xlxr(const float* __restrict__ xcat,
                            const float* __restrict__ Wl, const float* __restrict__ Wr,
                            int in_dim, int N, int ntiles,
                            float* __restrict__ xl, float* __restrict__ xr) {
    __shared__ __align__(32) _Float16 sWl[16 * 32];
    __shared__ __align__(32) _Float16 sWr[16 * 32];
    // cooperative zero-padded weight staging (once per block)
    for (int idx = threadIdx.x; idx < 16 * 32; idx += blockDim.x) {
        int n = idx >> 5, k = idx & 31;
        bool ok = (n < 10) && (k < in_dim);
        sWl[idx] = (_Float16)(ok ? Wl[n * in_dim + k] : 0.f);
        sWr[idx] = (_Float16)(ok ? Wr[n * in_dim + k] : 0.f);
    }
    __syncthreads();
    int gwave = (blockIdx.x * blockDim.x + threadIdx.x) >> 5;
    int lane  = threadIdx.x & 31;
    if (gwave >= ntiles) return;           // wave-uniform: EXEC all-1 for WMMA
    int hi  = lane >> 4;
    int l16 = lane & 15;
    int row = gwave * 16 + l16;
    if (row >= N) row = N - 1;             // clamp: valid rows unaffected (row-independent)
    v16h a = load_afrag(xcat, row, hi);
    // B fragments from LDS: N = lane%16, K = 16*hi + i
    int kbB = hi * 16;
    v16h bl = *(const v16h*)(&sWl[l16 * 32 + kbB]);
    v16h br = *(const v16h*)(&sWr[l16 * 32 + kbB]);
    v8f c0 = {}, c1 = {};
    c0 = __builtin_amdgcn_wmma_f32_16x16x32_f16(false, a, false, bl, (short)0, c0, false, false);
    c1 = __builtin_amdgcn_wmma_f32_16x16x32_f16(false, a, false, br, (short)0, c1, false, false);
    // C/D: col = lane%16; VGPR r -> row r (+8 for upper lane half)
    if (gwave * 16 + 16 <= N) {
#pragma unroll
        for (int r = 0; r < 8; ++r) {
            int rr = gwave * 16 + r + hi * 8;
            xl[(size_t)rr * 16 + l16] = c0[r];
            xr[(size_t)rr * 16 + l16] = c1[r];
        }
    } else {
#pragma unroll
        for (int r = 0; r < 8; ++r) {
            int rr = gwave * 16 + r + hi * 8;
            if (rr < N) {
                xl[(size_t)rr * 16 + l16] = c0[r];
                xr[(size_t)rr * 16 + l16] = c1[r];
            }
        }
    }
}

// ---------------- value head GEMM: Vn[N,32] = relu(xcat @ vt_W^T + vt_b), 20 valid cols ----
__global__ void k_gemm_vn(const float* __restrict__ xcat,
                          const float* __restrict__ W, const float* __restrict__ b,
                          int N, int ntiles, float* __restrict__ Vn) {
    __shared__ __align__(32) _Float16 sW[32 * 32];
    __shared__ float sB[32];
    for (int idx = threadIdx.x; idx < 32 * 32; idx += blockDim.x) {
        int n = idx >> 5, k = idx & 31;
        bool ok = (n < 20) && (k < 29);
        sW[idx] = (_Float16)(ok ? W[n * 29 + k] : 0.f);
    }
    if (threadIdx.x < 32) sB[threadIdx.x] = (threadIdx.x < 20) ? b[threadIdx.x] : 0.f;
    __syncthreads();
    int gwave = (blockIdx.x * blockDim.x + threadIdx.x) >> 5;
    int lane  = threadIdx.x & 31;
    if (gwave >= ntiles) return;
    int hi = lane >> 4, l16 = lane & 15;
    int row = gwave * 16 + l16;
    if (row >= N) row = N - 1;
    v16h a = load_afrag(xcat, row, hi);
    int kbB = hi * 16;
    int n0 = l16, n1 = 16 + l16;
    v16h b0 = *(const v16h*)(&sW[n0 * 32 + kbB]);
    v16h b1 = *(const v16h*)(&sW[n1 * 32 + kbB]);
    v8f c0 = {}, c1 = {};
    c0 = __builtin_amdgcn_wmma_f32_16x16x32_f16(false, a, false, b0, (short)0, c0, false, false);
    c1 = __builtin_amdgcn_wmma_f32_16x16x32_f16(false, a, false, b1, (short)0, c1, false, false);
    float bi0 = sB[n0], bi1 = sB[n1];
    if (gwave * 16 + 16 <= N) {
#pragma unroll
        for (int r = 0; r < 8; ++r) {
            int rr = gwave * 16 + r + hi * 8;
            Vn[(size_t)rr * 32 + n0] = fmaxf(c0[r] + bi0, 0.f);
            Vn[(size_t)rr * 32 + n1] = fmaxf(c1[r] + bi1, 0.f);   // cols >=20 are dead pad
        }
    } else {
#pragma unroll
        for (int r = 0; r < 8; ++r) {
            int rr = gwave * 16 + r + hi * 8;
            if (rr < N) {
                Vn[(size_t)rr * 32 + n0] = fmaxf(c0[r] + bi0, 0.f);
                Vn[(size_t)rr * 32 + n1] = fmaxf(c1[r] + bi1, 0.f);
            }
        }
    }
}

// ---------------- edge pass 1: e_i and segment max over dst ----------------
__global__ void k_edge1(const int* __restrict__ esrc, const int* __restrict__ edst,
                        int E, int EN,
                        const float* __restrict__ xl, const float* __restrict__ xr,
                        const float* __restrict__ att,
                        float* __restrict__ ework, unsigned* __restrict__ emaxu) {
    int i = blockIdx.x * blockDim.x + threadIdx.x;
    if (i >= EN) return;
    int s = (i < E) ? esrc[i] : (i - E);
    int d = (i < E) ? edst[i] : (i - E);
    const float4* ls4 = (const float4*)(xl + (size_t)s * 16);
    const float4* rd4 = (const float4*)(xr + (size_t)d * 16);
    float4 l0 = ls4[0], l1 = ls4[1], l2 = ls4[2];
    float4 r0 = rd4[0], r1 = rd4[1], r2 = rd4[2];
    float e = att[0] * leaky(l0.x + r0.x)
            + att[1] * leaky(l0.y + r0.y)
            + att[2] * leaky(l0.z + r0.z)
            + att[3] * leaky(l0.w + r0.w)
            + att[4] * leaky(l1.x + r1.x)
            + att[5] * leaky(l1.y + r1.y)
            + att[6] * leaky(l1.z + r1.z)
            + att[7] * leaky(l1.w + r1.w)
            + att[8] * leaky(l2.x + r2.x)
            + att[9] * leaky(l2.y + r2.y);
    ework[i] = e;
    atomicMax(&emaxu[d], fenc(e));
}

// ---------------- edge pass 2: exp + segment sum ----------------
__global__ void k_edge2(const int* __restrict__ edst, int E, int EN,
                        const unsigned* __restrict__ emaxu,
                        float* __restrict__ ework, float* __restrict__ denom) {
    int i = blockIdx.x * blockDim.x + threadIdx.x;
    if (i >= EN) return;
    int d = (i < E) ? edst[i] : (i - E);
    float ee = __expf(ework[i] - fdec(emaxu[d]));
    ework[i] = ee;
    atomicAdd(&denom[d], ee);
}

// ---------------- edge pass 3: weighted aggregation of xl[src] into xa[dst] ----------------
__global__ void k_edge3(const int* __restrict__ esrc, const int* __restrict__ edst,
                        int E, int EN,
                        const float* __restrict__ xl, const float* __restrict__ ework,
                        const float* __restrict__ denom, float* __restrict__ xa) {
    int i = blockIdx.x * blockDim.x + threadIdx.x;
    if (i >= EN) return;
    int s = (i < E) ? esrc[i] : (i - E);
    int d = (i < E) ? edst[i] : (i - E);
    float alpha = ework[i] / denom[d];
    const float4* ls4 = (const float4*)(xl + (size_t)s * 16);
    float4 l0 = ls4[0], l1 = ls4[1], l2 = ls4[2];
    float* od = xa + (size_t)d * 16;
    atomicAdd(&od[0], alpha * l0.x);
    atomicAdd(&od[1], alpha * l0.y);
    atomicAdd(&od[2], alpha * l0.z);
    atomicAdd(&od[3], alpha * l0.w);
    atomicAdd(&od[4], alpha * l1.x);
    atomicAdd(&od[5], alpha * l1.y);
    atomicAdd(&od[6], alpha * l1.z);
    atomicAdd(&od[7], alpha * l1.w);
    atomicAdd(&od[8], alpha * l2.x);
    atomicAdd(&od[9], alpha * l2.y);
}

// ---------------- finalize layer: bias + relu, build next concat ----------------
// mode 0: xcat = [x(10), x1(15), 0(7)] ; mode 1: xcat = [x(10), x1(15), x2(4), 0(3)]
__global__ void k_finalize(float* __restrict__ xa, const float* __restrict__ x1,
                           const float* __restrict__ x2, const float* __restrict__ b,
                           float* __restrict__ xcat, int N, int mode) {
    int n = blockIdx.x * blockDim.x + threadIdx.x;
    if (n >= N) return;
    float* xrow = xa + (size_t)n * 16;
    float* c    = xcat + (size_t)n * 32;
#pragma unroll
    for (int f = 0; f < 10; ++f) {
        float v = fmaxf(xrow[f] + b[f], 0.f);
        xrow[f] = v;
        c[f] = v;
    }
    const float* r1 = x1 + (size_t)n * 15;
#pragma unroll
    for (int f = 0; f < 15; ++f) c[10 + f] = r1[f];
    if (mode == 0) {
#pragma unroll
        for (int f = 25; f < 32; ++f) c[f] = 0.f;
    } else {
#pragma unroll
        for (int j = 0; j < 4; ++j) c[25 + j] = x2[j];
        c[29] = 0.f; c[30] = 0.f; c[31] = 0.f;
    }
}

// ---------------- value attention passes (Vn stride 32) ----------------
__global__ void k_vscore(const float* __restrict__ Vn, const float* __restrict__ vaW,
                         const float* __restrict__ vab, int N,
                         float* __restrict__ sw, unsigned* __restrict__ redm) {
    int n = blockIdx.x * blockDim.x + threadIdx.x;
    if (n >= N) return;
    const float* r = Vn + (size_t)n * 32;
    float s = vab[0];
#pragma unroll
    for (int f = 0; f < 20; ++f) s += vaW[f] * r[f];
    sw[n] = s;
    atomicMax(&redm[0], fenc(s));
}

__global__ void k_vsoft1(float* __restrict__ sw, const unsigned* __restrict__ redm,
                         int N, float* __restrict__ reds) {
    int n = blockIdx.x * blockDim.x + threadIdx.x;
    if (n >= N) return;
    float ee = __expf(sw[n] - fdec(redm[0]));
    sw[n] = ee;
    atomicAdd(&reds[0], ee);
}

__global__ void k_vsoft2(const float* __restrict__ sw, const float* __restrict__ reds,
                         const float* __restrict__ Vn, const float* __restrict__ vvW,
                         const float* __restrict__ vvb, int N, float* __restrict__ vp) {
    int n = blockIdx.x * blockDim.x + threadIdx.x;
    if (n >= N) return;
    float w = sw[n] / reds[0];
    const float* r = Vn + (size_t)n * 32;
#pragma unroll
    for (int f = 0; f < 10; ++f) {
        float v = vvb[f];
        const float* wr = vvW + f * 20;
#pragma unroll
        for (int g = 0; g < 20; ++g) v += wr[g] * r[g];
        atomicAdd(&vp[f], w * v);
    }
}

__global__ void k_vfinal(const float* __restrict__ vp, const float* __restrict__ vlW,
                         const float* __restrict__ vlb, float* __restrict__ out) {
    float acc = vlb[0];
#pragma unroll
    for (int f = 0; f < 10; ++f) acc += vlW[f] * fmaxf(vp[f], 0.f);
    out[0] = tanhf(acc);
}

// ---------------- move head: per (m,k) feat -> attn logit + value ----------------
__global__ void k_moves(const int* __restrict__ msrc, const int* __restrict__ mdst,
                        const float* __restrict__ marm, const int* __restrict__ iatk,
                        const float* __restrict__ xa, const float* __restrict__ x1,
                        const float* __restrict__ atW, const float* __restrict__ atb,
                        const float* __restrict__ dtW, const float* __restrict__ dtb,
                        const float* __restrict__ oaW, const float* __restrict__ oab,
                        const float* __restrict__ ovW, const float* __restrict__ ovb,
                        int MK, float* __restrict__ alog, float* __restrict__ pv) {
    int i = blockIdx.x * blockDim.x + threadIdx.x;
    if (i >= MK) return;
    int s = msrc[i], d = mdst[i];
    float arm = marm[i];
    int ia = iatk[i];
    const float* xs  = xa + (size_t)s * 16;
    const float* x1s = x1 + (size_t)s * 15;
    const float* x1d = x1 + (size_t)d * 15;
    float in[48];
    int len;
    const float *W, *bb;
    if (ia == 1) {
        len = 48; W = atW; bb = atb;
        const float* xd = xa + (size_t)d * 16;
#pragma unroll
        for (int f = 0; f < 10; ++f) in[f] = xs[f];
#pragma unroll
        for (int f = 0; f < 10; ++f) in[10 + f] = xd[f];
#pragma unroll
        for (int j = 0; j < 12; ++j) in[20 + j] = x1s[3 + j];
#pragma unroll
        for (int j = 0; j < 14; ++j) in[32 + j] = x1d[1 + j];
        in[46] = arm;
        in[47] = 0.6f * arm - 0.7f * (x1d[3] + x1d[4]);
    } else {
        len = 23; W = dtW; bb = dtb;
#pragma unroll
        for (int f = 0; f < 10; ++f) in[f] = xs[f];
#pragma unroll
        for (int j = 0; j < 12; ++j) in[10 + j] = x1s[3 + j];
        in[22] = arm;
    }
    float aacc = oab[0], vacc = ovb[0];
    for (int j = 0; j < 20; ++j) {
        float f = bb[j];
        const float* wr = W + j * len;
        for (int t = 0; t < len; ++t) f += wr[t] * in[t];
        f = fmaxf(f, 0.f);
        aacc += oaW[j] * f;
        vacc += ovW[j] * f;
    }
    alog[i] = aacc;
    pv[i]   = vacc;
}

// ---------------- per-move softmax over K=16 orders ----------------
__global__ void k_pmove(const float* __restrict__ alog, const float* __restrict__ pv,
                        int M, float* __restrict__ pm) {
    int m = blockIdx.x * blockDim.x + threadIdx.x;
    if (m >= M) return;
    const float* a = alog + (size_t)m * 16;
    const float* v = pv + (size_t)m * 16;
    float mx = a[0];
#pragma unroll
    for (int k = 1; k < 16; ++k) mx = fmaxf(mx, a[k]);
    float s = 0.f, acc = 0.f;
#pragma unroll
    for (int k = 0; k < 16; ++k) {
        float e = __expf(a[k] - mx);
        s += e;
        acc += e * v[k];
    }
    pm[m] = acc / s;
}

// ---------------- log_softmax over M moves ----------------
__global__ void k_logsoftmax(const float* __restrict__ pm, float* __restrict__ out, int M) {
    __shared__ float sh[1024];
    int t = threadIdx.x;
    float mx = -3.4e38f;
    for (int i = t; i < M; i += 1024) mx = fmaxf(mx, pm[i]);
    sh[t] = mx; __syncthreads();
    for (int s = 512; s > 0; s >>= 1) { if (t < s) sh[t] = fmaxf(sh[t], sh[t + s]); __syncthreads(); }
    float gmx = sh[0]; __syncthreads();
    float sm = 0.f;
    for (int i = t; i < M; i += 1024) sm += __expf(pm[i] - gmx);
    sh[t] = sm; __syncthreads();
    for (int s = 512; s > 0; s >>= 1) { if (t < s) sh[t] += sh[t + s]; __syncthreads(); }
    float lse = gmx + logf(sh[0]);
    for (int i = t; i < M; i += 1024) out[1 + i] = pm[i] - lse;
}

extern "C" void kernel_launch(void* const* d_in, const int* in_sizes, int n_in,
                              void* d_out, int out_size, void* d_ws, size_t ws_size,
                              hipStream_t stream) {
    const float* x1    = (const float*)d_in[0];
    const float* x2    = (const float*)d_in[1];
    const int*   edges = (const int*)d_in[2];
    const int*   msrc  = (const int*)d_in[3];
    const int*   mdst  = (const int*)d_in[4];
    const float* marm  = (const float*)d_in[5];
    const int*   iatk  = (const int*)d_in[6];
    const float* gWl[3]  = {(const float*)d_in[7],  (const float*)d_in[11], (const float*)d_in[15]};
    const float* gWr[3]  = {(const float*)d_in[8],  (const float*)d_in[12], (const float*)d_in[16]};
    const float* gatt[3] = {(const float*)d_in[9],  (const float*)d_in[13], (const float*)d_in[17]};
    const float* gb[3]   = {(const float*)d_in[10], (const float*)d_in[14], (const float*)d_in[18]};
    const float* vt_W = (const float*)d_in[19]; const float* vt_b = (const float*)d_in[20];
    const float* va_W = (const float*)d_in[21]; const float* va_b = (const float*)d_in[22];
    const float* vv_W = (const float*)d_in[23]; const float* vv_b = (const float*)d_in[24];
    const float* vl_W = (const float*)d_in[25]; const float* vl_b = (const float*)d_in[26];
    const float* at_W = (const float*)d_in[27]; const float* at_b = (const float*)d_in[28];
    const float* dt_W = (const float*)d_in[29]; const float* dt_b = (const float*)d_in[30];
    const float* oa_W = (const float*)d_in[31]; const float* oa_b = (const float*)d_in[32];
    const float* ov_W = (const float*)d_in[33]; const float* ov_b = (const float*)d_in[34];

    const int N  = in_sizes[0] / 15;
    const int E  = in_sizes[2] / 2;
    const int MK = in_sizes[3];
    const int M  = MK / 16;
    const int EN = E + N;

    // workspace layout (floats, 16-float aligned so float4 loads stay 16B aligned)
    auto alignup = [](size_t v) { return (v + 15) & ~(size_t)15; };
    float* ws = (float*)d_ws;
    size_t o = 0;
    float*    xl    = ws + o; o = alignup(o + (size_t)N * 16);
    float*    xr    = ws + o; o = alignup(o + (size_t)N * 16);
    float*    xa    = ws + o; o = alignup(o + (size_t)N * 16);   // }
    unsigned* emaxu = (unsigned*)(ws + o); o = alignup(o + N);   // } one contiguous zero fill
    float*    denom = ws + o; o = alignup(o + N);                // }
    size_t zfillN   = (size_t)(denom - xa) + N;                  // elements from xa..denom end
    float*    xcat  = ws + o; o = alignup(o + (size_t)N * 32);
    float*    ework = ws + o; o = alignup(o + (size_t)EN);
    float*    Vn    = ws + o; o = alignup(o + (size_t)N * 32);
    float*    alog  = ws + o; o = alignup(o + (size_t)MK);
    float*    pv    = ws + o; o = alignup(o + (size_t)MK);
    float*    pm    = ws + o; o = alignup(o + (size_t)M);
    unsigned* redm  = (unsigned*)(ws + o); o = alignup(o + 4);
    float*    reds  = ws + o; o = alignup(o + 4);
    float*    vp    = ws + o; o = alignup(o + 16);
    (void)ws_size; (void)n_in; (void)out_size;

    const int B = 256;
    auto cdiv = [](long a, long b) { return (int)((a + b - 1) / b); };
    const int ntiles     = (N + 15) / 16;
    const int gemmBlocks = cdiv(ntiles, B / 32);
    const int* esrc = edges;
    const int* edst = edges + E;
    float* out = (float*)d_out;

    // ---- GATv2 layers ----
    k_build_cat1<<<cdiv(N, B), B, 0, stream>>>(x1, xcat, N);
    const int indim[3] = {15, 25, 25};
    for (int L = 0; L < 3; ++L) {
        k_gemm_xlxr<<<gemmBlocks, B, 0, stream>>>(xcat, gWl[L], gWr[L], indim[L], N, ntiles, xl, xr);
        k_fill<<<cdiv((long)zfillN, B), B, 0, stream>>>((unsigned*)xa, 0u, (int)zfillN);
        k_edge1<<<cdiv(EN, B), B, 0, stream>>>(esrc, edst, E, EN, xl, xr, gatt[L], ework, emaxu);
        k_edge2<<<cdiv(EN, B), B, 0, stream>>>(edst, E, EN, emaxu, ework, denom);
        k_edge3<<<cdiv(EN, B), B, 0, stream>>>(esrc, edst, E, EN, xl, ework, denom, xa);
        k_finalize<<<cdiv(N, B), B, 0, stream>>>(xa, x1, x2, gb[L], xcat, N, (L < 2) ? 0 : 1);
    }

    // ---- value head ----
    k_gemm_vn<<<gemmBlocks, B, 0, stream>>>(xcat, vt_W, vt_b, N, ntiles, Vn);
    k_fill<<<1, 64, 0, stream>>>(redm, 0u, 1);
    k_fill<<<1, 64, 0, stream>>>((unsigned*)reds, 0u, 1);
    k_fill<<<1, 64, 0, stream>>>((unsigned*)vp, 0u, 16);
    k_vscore<<<cdiv(N, B), B, 0, stream>>>(Vn, va_W, va_b, N, ework, redm);
    k_vsoft1<<<cdiv(N, B), B, 0, stream>>>(ework, redm, N, reds);
    k_vsoft2<<<cdiv(N, B), B, 0, stream>>>(ework, reds, Vn, vv_W, vv_b, N, vp);
    k_vfinal<<<1, 1, 0, stream>>>(vp, vl_W, vl_b, out);

    // ---- move head ----
    k_moves<<<cdiv(MK, B), B, 0, stream>>>(msrc, mdst, marm, iatk, xa, x1,
                                           at_W, at_b, dt_W, dt_b, oa_W, oa_b, ov_W, ov_b,
                                           MK, alog, pv);
    k_pmove<<<cdiv(M, B), B, 0, stream>>>(alog, pv, M, pm);
    k_logsoftmax<<<1, 1024, 0, stream>>>(pm, out, M);
}
